// SelfAttention_55868934586521
// MI455X (gfx1250) — matrix operand order repaired and checked by reading, not compiled, not run on hardware
//
#include <hip/hip_runtime.h>
#include <hip/hip_bf16.h>

typedef __attribute__((ext_vector_type(16))) __bf16 v16bf;
typedef __attribute__((ext_vector_type(8)))  __bf16 v8bf;
typedef __attribute__((ext_vector_type(8)))  float  v8f;

#define NROWS 4096
#define DIM   1024

static __device__ __forceinline__ v16bf cat8(v8bf lo, v8bf hi) {
    return __builtin_shufflevector(lo, hi, 0,1,2,3,4,5,6,7,8,9,10,11,12,13,14,15);
}

// ---------------------------------------------------------------------------
// Kernel 1: C_bf16[4096x1024] = A_f32[4096x1024] @ W_f32[1024x1024]
// block tile 128x128, k-stage 64, 256 threads = 8 waves, wave tile 32x64.
// Double-buffered LDS staging: loads of stage s+1 overlap WMMAs of stage s.
// grid.z selects (W_K->Kb, W_Q->Qb, W_V->Vb)
// ---------------------------------------------------------------------------
#define AST 72    // LDS stride for A tile (elems), 144B: 16B aligned
#define BST 136   // LDS stride for B tile (elems), 272B: 16B aligned

static __device__ __forceinline__ void stage_proj(
    const float* __restrict__ X, const float* __restrict__ W,
    __bf16* __restrict__ As, __bf16* __restrict__ Bs,
    int tid, int mBase, int nBase, int kk)
{
    // stage A: 128 rows x 64 cols fp32 -> bf16 (2048 float4 chunks)
    #pragma unroll
    for (int i = 0; i < 8; ++i) {
        int chunk = tid + i * 256;
        int r  = chunk >> 4;          // 16 float4 per row
        int c4 = chunk & 15;
        float4 v = *(const float4*)(X + (size_t)(mBase + r) * DIM + kk + c4 * 4);
        __bf16* dst = &As[r * AST + c4 * 4];
        dst[0] = (__bf16)v.x; dst[1] = (__bf16)v.y;
        dst[2] = (__bf16)v.z; dst[3] = (__bf16)v.w;
    }
    // stage B: 64 rows x 128 cols fp32 -> bf16 (2048 float4 chunks)
    #pragma unroll
    for (int i = 0; i < 8; ++i) {
        int chunk = tid + i * 256;
        int r  = chunk >> 5;          // 32 float4 per row
        int c4 = chunk & 31;
        float4 v = *(const float4*)(W + (size_t)(kk + r) * DIM + nBase + c4 * 4);
        __bf16* dst = &Bs[r * BST + c4 * 4];
        dst[0] = (__bf16)v.x; dst[1] = (__bf16)v.y;
        dst[2] = (__bf16)v.z; dst[3] = (__bf16)v.w;
    }
}

__global__ __launch_bounds__(256) void proj_gemm_bf16(
    const float* __restrict__ X,
    const float* __restrict__ W0, const float* __restrict__ W1, const float* __restrict__ W2,
    __bf16* __restrict__ O0, __bf16* __restrict__ O1, __bf16* __restrict__ O2)
{
    const float* W = (blockIdx.z == 0) ? W0 : (blockIdx.z == 1 ? W1 : W2);
    __bf16*      O = (blockIdx.z == 0) ? O0 : (blockIdx.z == 1 ? O1 : O2);

    __shared__ __bf16 As[2][128 * AST];   // 2 x 18 KB
    __shared__ __bf16 Bs[2][64 * BST];    // 2 x 17 KB

    const int tid = threadIdx.x;
    const int w   = tid >> 5;
    const int L   = tid & 31;
    const int wm  = w & 3;        // 4 waves along M
    const int wn  = w >> 2;       // 2 waves along N
    const int h   = L >> 4;
    const int ml  = L & 15;
    const int mBase = blockIdx.x * 128;
    const int nBase = blockIdx.y * 128;

    v8f acc[2][4] = {};

    stage_proj(X, W, As[0], Bs[0], tid, mBase, nBase, 0);
    __syncthreads();

    for (int s = 0; s < DIM / 64; ++s) {
        const int buf = s & 1;
        // prefetch next stage into the other buffer (overlaps with WMMAs below)
        if (s + 1 < DIM / 64)
            stage_proj(X, W, As[buf ^ 1], Bs[buf ^ 1], tid, mBase, nBase, (s + 1) * 64);

        #pragma unroll
        for (int ki = 0; ki < 2; ++ki) {
            v16bf a[2], b[4];
            #pragma unroll
            for (int mt = 0; mt < 2; ++mt) {
                const __bf16* p = &As[buf][(wm * 32 + mt * 16 + ml) * AST + ki * 32 + h * 8];
                a[mt] = cat8(*(const v8bf*)p, *(const v8bf*)(p + 16));
            }
            #pragma unroll
            for (int nt = 0; nt < 4; ++nt) {
                const __bf16* p = &Bs[buf][(ki * 32 + h * 8) * BST + wn * 64 + nt * 16 + ml];
                #pragma unroll
                for (int e = 0; e < 16; ++e) b[nt][e] = p[((e & 7) + (e >> 3) * 16) * BST];
            }
            #pragma unroll
            for (int mt = 0; mt < 2; ++mt)
                #pragma unroll
                for (int nt = 0; nt < 4; ++nt)
                    acc[mt][nt] = __builtin_amdgcn_wmma_f32_16x16x32_bf16(
                        false, a[mt], false, b[nt], (short)0, acc[mt][nt], false, false);
        }
        __syncthreads();
    }

    // epilogue: f32 acc -> bf16, row-major store
    #pragma unroll
    for (int mt = 0; mt < 2; ++mt)
        #pragma unroll
        for (int nt = 0; nt < 4; ++nt) {
            int col = nBase + wn * 64 + nt * 16 + ml;
            int rb  = mBase + wm * 32 + mt * 16 + h * 8;
            #pragma unroll
            for (int g = 0; g < 8; ++g)
                O[(size_t)(rb + g) * DIM + col] = (__bf16)acc[mt][nt][g];
        }
}

// ---------------------------------------------------------------------------
// Kernel 2: flash attention. 16 query rows / workgroup, 8 waves, 128 keys/iter.
// Wave w: computes S^T tile for its 16 keys (full D reduction), online
// softmax stats, then accumulates O for V-columns [w*128, w*128+128).
// K and V operands load straight from global (L2-resident, 16 MB vs 192 MB L2).
// ---------------------------------------------------------------------------
#define QTS 24    // Qt LDS stride (elems), 48B: 16B aligned
#define PST 136   // P  LDS stride (elems), 272B: 16B aligned

__global__ __launch_bounds__(256) void flash_attn_bf16(
    const __bf16* __restrict__ Qb, const __bf16* __restrict__ Kb,
    const __bf16* __restrict__ Vb, float* __restrict__ out)
{
    __shared__ __bf16 Qt[DIM * QTS];       // Q^T: [feat][query], 48 KB
    __shared__ __bf16 Pl[16 * PST];        // P:   [query][128 keys]
    __shared__ float bmaxS[8 * 16], bsumS[8 * 16], fbS[8 * 16];
    __shared__ float foldS[16], mrun[16], lrun[16];

    const int tid = threadIdx.x;
    const int w   = tid >> 5;
    const int L   = tid & 31;
    const int h   = L >> 4;
    const int ml  = L & 15;
    const int qBase = blockIdx.x * 16;

    // load + transpose this block's 16 Q rows into Qt[feat][query]
    #pragma unroll
    for (int i = 0; i < 8; ++i) {
        int chunk = tid + i * 256;          // 16 rows * 128 chunks of 8
        int r  = chunk >> 7;
        int c8 = chunk & 127;
        v8bf v = *(const v8bf*)(Qb + (size_t)(qBase + r) * DIM + c8 * 8);
        #pragma unroll
        for (int e = 0; e < 8; ++e)
            Qt[(c8 * 8 + e) * QTS + r] = v[e];
    }
    if (tid < 16) { mrun[tid] = -1e30f; lrun[tid] = 0.0f; }
    __syncthreads();

    v8f o[8] = {};
    const float scale = 0.03125f;          // 1/sqrt(1024)
    const float L2E   = 1.44269504f;

    for (int j = 0; j < NROWS / 128; ++j) {
        const int jBase = j * 128;
        const int kb    = jBase + w * 16;  // this wave's 16 keys

        // ---- S^T tile (16 keys x 16 queries), K operand straight from global
        v8f s = {};
        #pragma unroll 4
        for (int kt = 0; kt < 32; ++kt) {
            const __bf16* kp = Kb + (size_t)(kb + ml) * DIM + kt * 32 + h * 8;
            v16bf a = cat8(*(const v8bf*)kp, *(const v8bf*)(kp + 16));
            const __bf16* qp = &Qt[(kt * 32 + h * 8) * QTS + ml];
            v16bf b;
            #pragma unroll
            for (int e = 0; e < 16; ++e) b[e] = qp[((e & 7) + (e >> 3) * 16) * QTS];
            s = __builtin_amdgcn_wmma_f32_16x16x32_bf16(false, a, false, b, (short)0, s, false, false);
        }

        // ---- block softmax stats (keys of query q live in lanes q and q+16)
        float sv[8], p[8];
        float bm = -1e30f;
        #pragma unroll
        for (int g = 0; g < 8; ++g) { sv[g] = s[g] * scale; bm = fmaxf(bm, sv[g]); }
        bm = fmaxf(bm, __shfl_xor(bm, 16, 32));
        float bs = 0.0f;
        #pragma unroll
        for (int g = 0; g < 8; ++g) { p[g] = exp2f((sv[g] - bm) * L2E); bs += p[g]; }
        bs += __shfl_xor(bs, 16, 32);
        if (h == 0) { bmaxS[w * 16 + ml] = bm; bsumS[w * 16 + ml] = bs; }
        __syncthreads();

        // ---- merge the 8 key-block stats (one lane per query)
        if (tid < 16) {
            int q = tid;
            float mo = mrun[q], mn = mo;
            #pragma unroll
            for (int b = 0; b < 8; ++b) mn = fmaxf(mn, bmaxS[b * 16 + q]);
            float fo = exp2f((mo - mn) * L2E);
            float l  = lrun[q] * fo;
            #pragma unroll
            for (int b = 0; b < 8; ++b) {
                float f = exp2f((bmaxS[b * 16 + q] - mn) * L2E);
                fbS[b * 16 + q] = f;
                l += bsumS[b * 16 + q] * f;
            }
            mrun[q] = mn; lrun[q] = l; foldS[q] = fo;
        }
        __syncthreads();

        // ---- write rescaled bf16 P, rescale running O
        float fb = fbS[w * 16 + ml];
        #pragma unroll
        for (int g = 0; g < 8; ++g)
            Pl[ml * PST + w * 16 + h * 8 + g] = (__bf16)(p[g] * fb);
        float fold[8];
        #pragma unroll
        for (int g = 0; g < 8; ++g) fold[g] = foldS[h * 8 + g];
        #pragma unroll
        for (int nt = 0; nt < 8; ++nt)
            #pragma unroll
            for (int g = 0; g < 8; ++g) o[nt][g] *= fold[g];
        __syncthreads();

        // ---- O += P @ V  (V operand straight from global; hot in L2)
        #pragma unroll
        for (int kt = 0; kt < 4; ++kt) {
            const __bf16* pp = &Pl[ml * PST + kt * 32 + h * 8];
            v16bf a = cat8(*(const v8bf*)pp, *(const v8bf*)(pp + 16));
            #pragma unroll
            for (int nt = 0; nt < 8; ++nt) {
                const __bf16* vp = Vb + (size_t)(jBase + kt * 32 + h * 8) * DIM + w * 128 + nt * 16 + ml;
                v16bf b;
                #pragma unroll
                for (int e = 0; e < 16; ++e) b[e] = vp[(size_t)((e & 7) + (e >> 3) * 16) * DIM];
                o[nt] = __builtin_amdgcn_wmma_f32_16x16x32_bf16(false, a, false, b, (short)0, o[nt], false, false);
            }
        }
    }

    // ---- finalize: divide by softmax denominator, fp32 store
    float linv[8];
    #pragma unroll
    for (int g = 0; g < 8; ++g) linv[g] = 1.0f / lrun[h * 8 + g];
    #pragma unroll
    for (int nt = 0; nt < 8; ++nt)
        #pragma unroll
        for (int g = 0; g < 8; ++g)
            out[(size_t)(qBase + h * 8 + g) * DIM + w * 128 + nt * 16 + ml] = o[nt][g] * linv[g];
}

// ---------------------------------------------------------------------------
extern "C" void kernel_launch(void* const* d_in, const int* in_sizes, int n_in,
                              void* d_out, int out_size, void* d_ws, size_t ws_size,
                              hipStream_t stream) {
    const float* X   = (const float*)d_in[0];
    const float* W_K = (const float*)d_in[1];
    const float* W_Q = (const float*)d_in[2];
    const float* W_V = (const float*)d_in[3];
    float* out = (float*)d_out;

    __bf16* Kb = (__bf16*)d_ws;                       // 8 MB
    __bf16* Qb = Kb + (size_t)NROWS * DIM;            // 8 MB
    __bf16* Vb = Qb + (size_t)NROWS * DIM;            // 8 MB

    dim3 gridP(NROWS / 128, DIM / 128, 3);
    proj_gemm_bf16<<<gridP, 256, 0, stream>>>(X, W_K, W_Q, W_V, Kb, Qb, Vb);
    flash_attn_bf16<<<NROWS / 16, 256, 0, stream>>>(Qb, Kb, Vb, out);
}
